// VisionLanguage2DSemanticMapModule_28071906246834
// MI455X (gfx1250) — matrix operand order repaired and checked by reading, not compiled, not run on hardware
//
#include <hip/hip_runtime.h>
#include <hip/hip_bf16.h>
#include <math.h>

// ---------- CDNA5 vector types ----------
typedef _Float16 v16h __attribute__((ext_vector_type(16)));
typedef float    v8f  __attribute__((ext_vector_type(8)));
typedef unsigned int v4u __attribute__((ext_vector_type(4)));
typedef int      v4i  __attribute__((ext_vector_type(4)));
typedef int      v8i  __attribute__((ext_vector_type(8)));

// ---------- problem constants ----------
#define FRAME_H 240
#define FRAME_W 320
#define PH      60          // FRAME_H / DU
#define PW      80          // FRAME_W / DU
#define NPTS    4800        // PH*PW
#define FDIM    128
#define NMAPC   133
#define VRR     64
#define NVH     80
#define LOCALM  240
#define GLOBALM 480
#define MIN_MH_ 13
#define MAX_MH_ 25
#define LM2     (LOCALM*LOCALM)        // 57600
#define GM2     (GLOBALM*GLOBALM)      // 230400

__device__ __forceinline__ int iabs_(int v){ return v < 0 ? -v : v; }
__device__ __forceinline__ int imin_(int a,int b){ return a<b?a:b; }
__device__ __forceinline__ int imax_(int a,int b){ return a>b?a:b; }
__device__ __forceinline__ float clamp01(float v){ return fminf(fmaxf(v,0.f),1.f); }

// ---------- tiny state kernels ----------
__global__ void k_init_state(float* st, int* lmb, const float* lp, const float* gp,
                             const int* lmb0, const float* org) {
  if (threadIdx.x == 0 && blockIdx.x == 0) {
    st[0]=lp[0]; st[1]=lp[1]; st[2]=lp[2];
    st[3]=gp[0]; st[4]=gp[1]; st[5]=gp[2];
    st[6]=org[0]; st[7]=org[1]; st[8]=org[2];
    lmb[0]=lmb0[0]; lmb[1]=lmb0[1]; lmb[2]=lmb0[2]; lmb[3]=lmb0[3];
  }
}

__global__ void k_reset_maps(const unsigned char* done, float* localMap, float* globalMap) {
  if (!*done) return;
  size_t i = (size_t)blockIdx.x * blockDim.x + threadIdx.x;
  if (i < (size_t)NMAPC * LM2) localMap[i] = 0.f;
  if (i < (size_t)NMAPC * GM2) globalMap[i] = 0.f;
}

__global__ void k_pose(const unsigned char* done, const float* delta, float* st, int* lmb) {
  if (threadIdx.x != 0 || blockIdx.x != 0) return;
  if (*done) {               // reset to centered state (recenter of zero map at (12,12,0))
    st[0]=6.f; st[1]=6.f; st[2]=0.f;
    st[3]=12.f; st[4]=12.f; st[5]=0.f;
    st[6]=6.f; st[7]=6.f; st[8]=0.f;
    lmb[0]=120; lmb[1]=360; lmb[2]=120; lmb[3]=360;
  }
  float x=st[0], y=st[1], o=st[2];
  float r = o * (float)(M_PI/180.0);
  y += delta[0]*sinf(r) + delta[1]*cosf(r);
  x += delta[0]*cosf(r) - delta[1]*sinf(r);
  o += delta[2] * 57.29577951308232f;
  o = fmodf(o - 180.f, 360.f) + 180.f;
  o = fmodf(o + 180.f, 360.f) - 180.f;
  st[0]=x; st[1]=y; st[2]=o;
}

// ---------- 4x4 average pooling via WMMA ----------
// pooled(60x80) = P(60x240) @ H(240x80) per channel; P entries are 1/16 (exact in f16).
// Stage A: horizontal 4-sums -> LDS (64 rows at a time, f16).
// Stage B: v_wmma_f32_16x16x32_f16, A = constant banded P tile, B = H tile from LDS.
__global__ void k_pool(const float* __restrict__ pix, float* __restrict__ pooled) {
  __shared__ _Float16 Hs[64 * PW];           // 10,240 B
  const int c = blockIdx.x;
  const float* plane = pix + (size_t)c * FRAME_H * FRAME_W;
  const int t = threadIdx.x;
  const int lane = t & 31, wave = t >> 5;
  const int m = lane & 15, hi = lane >> 4;

  // constant A tiles (16x32 f16, CDNA5 A-layout): a[ks][m,k] = 1/16 iff m == 8*ks + k/4
  v16h a0, a1;
#pragma unroll
  for (int h = 0; h < 16; ++h) {
    int v = h >> 1, p = h & 1;
    int kl = ((v < 4) ? 0 : 16) + (hi ? 8 : 0) + ((v & 3) << 1) + p;
    a0[h] = (_Float16)((m ==      (kl >> 2)) ? 0.0625f : 0.0f);
    a1[h] = (_Float16)((m == 8 + (kl >> 2)) ? 0.0625f : 0.0f);
  }

  for (int mt = 0; mt < 4; ++mt) {
    // stage A: rows [64*mt, 64*mt+64) of H into LDS (zero-pad past 240)
    for (int i = t; i < 64 * PW; i += 256) {
      int r = i / PW, xo = i % PW;
      int yg = mt * 64 + r;
      float s = 0.f;
      if (yg < FRAME_H) {
        const float* p4 = plane + (size_t)yg * FRAME_W + xo * 4;
        s = p4[0] + p4[1] + p4[2] + p4[3];
      }
      Hs[i] = (_Float16)s;
    }
    __syncthreads();
    if (wave < 5) {                          // 5 N-tiles of 16 cols
      const int nt = wave;
      const int xo = nt * 16 + m;
      v16h b0, b1;                            // B (32x16 f16) from LDS
#pragma unroll
      for (int h = 0; h < 16; ++h) {
        int v = h >> 1, p = h & 1;
        int kb = ((v < 4) ? 0 : 16) + (hi ? 8 : 0) + ((v & 3) << 1) + p;
        b0[h] = Hs[kb * PW + xo];
        b1[h] = Hs[(32 + kb) * PW + xo];
      }
      v8f acc = {};
      acc = __builtin_amdgcn_wmma_f32_16x16x32_f16(false, a0, false, b0, (short)0, acc, false, false);
      acc = __builtin_amdgcn_wmma_f32_16x16x32_f16(false, a1, false, b1, (short)0, acc, false, false);
#pragma unroll
      for (int rI = 0; rI < 8; ++rI) {
        int yo = mt * 16 + rI + (hi ? 8 : 0);
        if (yo < PH) pooled[(size_t)c * NPTS + yo * PW + xo] = acc[rI];
      }
    }
    __syncthreads();
  }
}

// ---------- trilinear splat (wave-per-point, channels on lanes) ----------
__global__ void k_splat(const float* __restrict__ depth, const float* __restrict__ pooled,
                        float* __restrict__ vox, int c0, int nc) {
  int gt = blockIdx.x * blockDim.x + threadIdx.x;
  int lane = gt & 31;
  int wave = gt >> 5;
  int wpp = (nc + 31) >> 5;
  int pt = wave / wpp;
  int cw = wave % wpp;
  if (pt >= NPTS) return;
  int pi = pt / PW, pj = pt % PW;
  float d  = depth[(size_t)(pi * 4) * FRAME_W + pj * 4];
  float gx = (float)(pj * 4);
  float gz = (float)(FRAME_H - 1 - pi * 4);
  float camF = 160.0f / tanf(0.68944854f);      // HFOV/2 = 39.5 deg
  float X = (gx - 159.5f) * d / camF + 160.0f;  // + SHIFT_X
  float Z = (gz - 119.5f) * d / camF + 88.0f;   // + AGENT_H
  float pos0 = X * 0.2f, pos1 = d * 0.2f, pos2 = Z * 0.2f + 8.0f;
  float f0 = floorf(pos0), f1 = floorf(pos1), f2 = floorf(pos2);
  float w0[2], w1[2], w2[2]; int p0[2], p1[2], p2[2];
#pragma unroll
  for (int ix = 0; ix < 2; ++ix) {
    float q;
    q = f0 + ix; { bool s = (q > 0.f) && (q < 64.f); w0[ix] = s ? (1.f - fabsf(pos0 - q)) : 0.f; p0[ix] = s ? (int)q : 0; }
    q = f1 + ix; { bool s = (q > 0.f) && (q < 64.f); w1[ix] = s ? (1.f - fabsf(pos1 - q)) : 0.f; p1[ix] = s ? (int)q : 0; }
    q = f2 + ix; { bool s = (q > 0.f) && (q < 80.f); w2[ix] = s ? (1.f - fabsf(pos2 - q)) : 0.f; p2[ix] = s ? (int)q : 0; }
  }
  int c = c0 + cw * 32 + lane;
  if (c >= c0 + nc || c >= 129) return;
  float feat = (c == 0) ? 1.0f : pooled[(size_t)(c - 1) * NPTS + pt];
  float* vb = vox + (size_t)(c - c0) * VRR * VRR * NVH;
#pragma unroll
  for (int a = 0; a < 2; ++a)
#pragma unroll
    for (int b = 0; b < 2; ++b)
#pragma unroll
      for (int e = 0; e < 2; ++e) {
        float w = w0[a] * w1[b] * w2[e];
        if (w != 0.f)
          atomicAdd(&vb[((size_t)p0[a] * VRR + p1[b]) * NVH + p2[e]], feat * w);
      }
}

// ---------- per-voxel round + height reductions ----------
__global__ void k_reduce(const float* __restrict__ vox, float* __restrict__ agent,
                         float* __restrict__ all0, int c0, int nc) {
  int i = blockIdx.x * blockDim.x + threadIdx.x;
  if (i >= nc * VRR * VRR) return;
  int cl = i / (VRR * VRR);
  int yx = i % (VRR * VRR);
  int y = yx / VRR, x = yx % VRR;
  const float* b = vox + ((size_t)cl * VRR * VRR + (size_t)x * VRR + y) * NVH;
  float sa = 0.f;
#pragma unroll
  for (int h = MIN_MH_; h < MAX_MH_; ++h) sa += rintf(b[h]);
  agent[(size_t)(c0 + cl) * VRR * VRR + yx] = sa;
  if (c0 + cl == 0) {
    float stt = 0.f;
    for (int h = 0; h < NVH; ++h) stt += rintf(b[h]);
    all0[yx] = stt;
  }
}

// ---------- assemble egocentric view ----------
__global__ void k_av(const float* __restrict__ agent, const float* __restrict__ all0,
                     float* __restrict__ av) {
  size_t i = (size_t)blockIdx.x * blockDim.x + threadIdx.x;
  if (i >= (size_t)NMAPC * LM2) return;
  int c = (int)(i / LM2);
  int r = (int)(i % LM2);
  int y = r / LOCALM, x = r % LOCALM;
  float val = 0.f;
  int yy = y - 120, xx = x - 88;
  if (yy >= 0 && yy < 64 && xx >= 0 && xx < 64) {
    int o = yy * 64 + xx;
    if (c == 0)       val = clamp01(agent[o]);
    else if (c == 1)  val = clamp01(all0[o]);
    else if (c >= 5)  val = agent[(size_t)(c - 4) * 4096 + o];
  }
  av[i] = val;
}

// ---------- rotation grid_sample: TDM-staged LDS tiles ----------
__global__ void k_rot(const float* __restrict__ av, const float* __restrict__ st,
                      float* __restrict__ rot) {
  __shared__ float tile[44 * 44];
  const int c = blockIdx.z, ty = blockIdx.y, tx = blockIdx.x;
  const float o = st[2];
  const float sto = (90.f - o) * (float)(M_PI / 180.0);
  const float cs = cosf(sto), sn = sinf(sto);
  const float kk = 2.0f / 239.0f;
  // sample coords are affine in (x,y):
  const float Ax = 119.5f * cs * kk, Ay = -119.5f * sn * kk, A0 = 119.5f * (1.f - cs + sn);
  const float Bx = 119.5f * sn * kk, By =  119.5f * cs * kk, B0 = 119.5f * (1.f - sn - cs);
  // bounding box of the tile's sample points (corners suffice, affine map)
  const float x0p = (float)(tx * 16), x1p = (float)(tx * 16 + 15);
  const float y0p = (float)(ty * 16), y1p = (float)(ty * 16 + 15);
  float sx00 = Ax * x0p + Ay * y0p + A0, sx10 = Ax * x1p + Ay * y0p + A0;
  float sx01 = Ax * x0p + Ay * y1p + A0, sx11 = Ax * x1p + Ay * y1p + A0;
  float sy00 = Bx * x0p + By * y0p + B0, sy10 = Bx * x1p + By * y0p + B0;
  float sy01 = Bx * x0p + By * y1p + B0, sy11 = Bx * x1p + By * y1p + B0;
  float mnx = fminf(fminf(sx00, sx10), fminf(sx01, sx11));
  float mxx = fmaxf(fmaxf(sx00, sx10), fmaxf(sx01, sx11));
  float mny = fminf(fminf(sy00, sy10), fminf(sy01, sy11));
  float mxy = fmaxf(fmaxf(sy00, sy10), fmaxf(sy01, sy11));
  int bx0 = imax_(0, imin_(239, (int)floorf(mnx)));
  int bx1 = imax_(bx0, imin_(239, (int)floorf(mxx) + 1));
  int by0 = imax_(0, imin_(239, (int)floorf(mny)));
  int by1 = imax_(by0, imin_(239, (int)floorf(mxy) + 1));
  int bw = imin_(bx1 - bx0 + 1, 44);
  int bh = imin_(by1 - by0 + 1, 44);

#if __has_builtin(__builtin_amdgcn_tensor_load_to_lds)
  if (threadIdx.x == 0) {
    unsigned long long ga =
        (unsigned long long)(const void*)(av + (size_t)c * LM2 + (size_t)by0 * LOCALM + bx0);
    v4u g0;
    g0[0] = 1u;                                   // count=1, user descriptor
    g0[1] = 0u;                                   // lds_addr: 'tile' is sole LDS alloc (offset 0)
    g0[2] = (unsigned)(ga & 0xFFFFFFFFull);       // global_addr[31:0]
    g0[3] = (unsigned)((ga >> 32) & 0x1FFFFFFull) | (2u << 30);  // addr[56:32] | type=2
    unsigned td0 = (unsigned)(LOCALM - bx0);      // remaining extent for OOB
    unsigned td1 = (unsigned)(LOCALM - by0);
    v8i g1;
    g1[0] = (int)(2u << 16);                      // workgroup_mask=0, data_size=4B
    g1[1] = (int)((td0 & 0xFFFFu) << 16);                                 // tensor_dim0 lo
    g1[2] = (int)(((td0 >> 16) & 0xFFFFu) | ((td1 & 0xFFFFu) << 16));     // dim0 hi | dim1 lo
    g1[3] = (int)(((td1 >> 16) & 0xFFFFu) | (((unsigned)bw & 0xFFFFu) << 16)); // dim1 hi | tile_dim0
    g1[4] = (int)((unsigned)bh & 0xFFFFu);        // tile_dim1 | tile_dim2=0
    g1[5] = (int)LOCALM;                          // tensor_dim0_stride lo
    g1[6] = 0;
    g1[7] = 0;
    v4i gz = {0, 0, 0, 0};
#if defined(__clang_major__) && (__clang_major__ >= 23)
    v8i gz8 = {0, 0, 0, 0, 0, 0, 0, 0};
    __builtin_amdgcn_tensor_load_to_lds(g0, g1, gz, gz, gz8, 0);
#else
    __builtin_amdgcn_tensor_load_to_lds(g0, g1, gz, gz, 0);
#endif
  }
#if __has_builtin(__builtin_amdgcn_s_wait_tensorcnt)
  __builtin_amdgcn_s_wait_tensorcnt(0);
#endif
  __syncthreads();
#else
  for (int i = threadIdx.x; i < bw * bh; i += 256) {
    int ry = i / bw, rx = i % bw;
    tile[i] = av[(size_t)c * LM2 + (size_t)(by0 + ry) * LOCALM + (bx0 + rx)];
  }
  __syncthreads();
#endif

  const int px = tx * 16 + (threadIdx.x & 15);
  const int py = ty * 16 + (threadIdx.x >> 4);
  float sx = Ax * px + Ay * py + A0;
  float sy = Bx * px + By * py + B0;
  float xf = floorf(sx), yf = floorf(sy);
  float acc = 0.f;
#pragma unroll
  for (int dy = 0; dy < 2; ++dy)
#pragma unroll
    for (int dx = 0; dx < 2; ++dx) {
      float xi = xf + dx, yi = yf + dy;
      if (xi >= 0.f && xi <= 239.f && yi >= 0.f && yi <= 239.f) {
        float w = (1.f - fabsf(sx - xi)) * (1.f - fabsf(sy - yi));
        int lx = (int)xi - bx0, ly = (int)yi - by0;
        if (lx >= 0 && lx < bw && ly >= 0 && ly < bh) acc += w * tile[ly * bw + lx];
      }
    }
  rot[(size_t)c * LM2 + (size_t)py * LOCALM + px] = acc;
}

// ---------- translation grid_sample + max-merge + agent mask ----------
__global__ void k_trn(const float* __restrict__ rot, const float* __restrict__ st,
                      float* __restrict__ localMap) {
  size_t i = (size_t)blockIdx.x * blockDim.x + threadIdx.x;
  if (i >= (size_t)NMAPC * LM2) return;
  int c = (int)(i / LM2);
  int r = (int)(i % LM2);
  int y = r / LOCALM, x = r % LOCALM;
  float p0 = st[0], p1 = st[1];
  float stx = -(p0 * 20.f - 120.f) / 120.f;
  float sty = -(p1 * 20.f - 120.f) / 120.f;
  float sx = (float)x + 119.5f * stx;
  float sy = (float)y + 119.5f * sty;
  float xf = floorf(sx), yf = floorf(sy);
  float acc = 0.f;
  const float* pc = rot + (size_t)c * LM2;
#pragma unroll
  for (int dy = 0; dy < 2; ++dy)
#pragma unroll
    for (int dx = 0; dx < 2; ++dx) {
      float xi = xf + dx, yi = yf + dy;
      if (xi >= 0.f && xi <= 239.f && yi >= 0.f && yi <= 239.f) {
        float w = (1.f - fabsf(sx - xi)) * (1.f - fabsf(sy - yi));
        acc += w * pc[(size_t)(int)yi * LOCALM + (int)xi];
      }
    }
  float m = fmaxf(localMap[i], acc);
  int cx = (int)(p0 * 20.f), cy = (int)(p1 * 20.f);
  float mask = (iabs_(y - cy) <= 2 && iabs_(x - cx) <= 2) ? 1.f : 0.f;
  if (c == 2) m = mask;
  else if (c == 3) m = fmaxf(m, mask);
  localMap[i] = m;
}

// ---------- conditional global update / recenter ----------
__global__ void k_gwrite(const unsigned char* upd, const int* lmb,
                         const float* __restrict__ localMap, float* __restrict__ globalMap) {
  if (!*upd) return;
  size_t i = (size_t)blockIdx.x * blockDim.x + threadIdx.x;
  if (i >= (size_t)NMAPC * LM2) return;
  int c = (int)(i / LM2);
  int r = (int)(i % LM2);
  int y = r / LOCALM, x = r % LOCALM;
  globalMap[(size_t)c * GM2 + (size_t)(lmb[0] + y) * GLOBALM + (lmb[2] + x)] = localMap[i];
}

__global__ void k_recenter(const unsigned char* upd, float* st, int* lmb,
                           float* outLP, float* outGP, float* outLMB, float* outORG) {
  if (threadIdx.x != 0 || blockIdx.x != 0) return;
  float gx = st[0] + st[6];
  float gy = st[1] + st[7];
  float go = st[2] + st[8];
  if (*upd) {
    int loc_r = (int)(gy * 20.f);
    int loc_c = (int)(gx * 20.f);
    int gx1 = imin_(imax_(loc_r - 120, 0), 240);
    int gy1 = imin_(imax_(loc_c - 120, 0), 240);
    lmb[0] = gx1; lmb[1] = gx1 + 240; lmb[2] = gy1; lmb[3] = gy1 + 240;
    st[6] = gy1 * 0.05f; st[7] = gx1 * 0.05f; st[8] = 0.f;
    st[3] = gx; st[4] = gy; st[5] = go;
    st[0] = gx - st[6]; st[1] = gy - st[7]; st[2] = go - st[8];
  }
  outLP[0]=st[0]; outLP[1]=st[1]; outLP[2]=st[2];
  outGP[0]=st[3]; outGP[1]=st[4]; outGP[2]=st[5];
  outLMB[0]=(float)lmb[0]; outLMB[1]=(float)lmb[1]; outLMB[2]=(float)lmb[2]; outLMB[3]=(float)lmb[3];
  outORG[0]=st[6]; outORG[1]=st[7]; outORG[2]=st[8];
}

__global__ void k_lfromg(const unsigned char* upd, const int* lmb,
                         const float* __restrict__ globalMap, float* __restrict__ localMap) {
  if (!*upd) return;
  size_t i = (size_t)blockIdx.x * blockDim.x + threadIdx.x;
  if (i >= (size_t)NMAPC * LM2) return;
  int c = (int)(i / LM2);
  int r = (int)(i % LM2);
  int y = r / LOCALM, x = r % LOCALM;
  localMap[i] = globalMap[(size_t)c * GM2 + (size_t)(lmb[0] + y) * GLOBALM + (lmb[2] + x)];
}

// ---------- map features ----------
__global__ void k_mf(const float* __restrict__ localMap, const float* __restrict__ globalMap,
                     float* __restrict__ mf) {
  size_t i = (size_t)blockIdx.x * blockDim.x + threadIdx.x;
  if (i >= (size_t)136 * LM2) return;
  int c = (int)(i / LM2);
  int r = (int)(i % LM2);
  int y = r / LOCALM, x = r % LOCALM;
  float v;
  if (c < 4) v = localMap[(size_t)c * LM2 + r];
  else if (c < 8) {
    const float* g = globalMap + (size_t)(c - 4) * GM2;
    int Y = 2 * y, X = 2 * x;
    v = fmaxf(fmaxf(g[(size_t)Y * GLOBALM + X],     g[(size_t)Y * GLOBALM + X + 1]),
              fmaxf(g[(size_t)(Y+1) * GLOBALM + X], g[(size_t)(Y+1) * GLOBALM + X + 1]));
  } else v = localMap[(size_t)(c - 3) * LM2 + r];
  mf[i] = v;
}

// =====================================================================
extern "C" void kernel_launch(void* const* d_in, const int* in_sizes, int n_in,
                              void* d_out, int out_size, void* d_ws, size_t ws_size,
                              hipStream_t stream) {
  (void)in_sizes; (void)n_in; (void)out_size; (void)ws_size;
  const float* obs   = (const float*)d_in[0];
  const float* pixF  = (const float*)d_in[1];
  const float* delta = (const float*)d_in[2];
  const unsigned char* dones = (const unsigned char*)d_in[3];
  const unsigned char* upds  = (const unsigned char*)d_in[4];

  float* out = (float*)d_out;
  float* wsf = (float*)d_ws;

  // workspace layout (float elements)
  float* pooled = wsf;                         // 614,400
  float* agent  = wsf + 614400;                // 129*4096
  float* all0   = wsf + 1142784;               // 4096
  float* av     = wsf + 1146880;               // 133*57600
  float* rotb   = wsf + 8807680;               // 133*57600
  float* vox    = wsf + 16468480;              // 43*327680
  float* st     = wsf + 30558720;              // 16 floats
  int*   lmbst  = (int*)(wsf + 30558736);      // 4 ints

  // output layout (float elements)
  float* outMF  = out;                         // (1,8,136,240,240)
  float* outL   = out + 62668800;              // (1,133,240,240)
  float* outG   = out + 70329600;              // (1,133,480,480)
  float* outLP  = out + 100972800;             // (1,8,3)
  float* outGP  = out + 100972824;             // (1,8,3)
  float* outLMB = out + 100972848;             // (1,8,4)
  float* outORG = out + 100972880;             // (1,8,3)

  // init carry state (fully rewritten every call -> deterministic under graph replay)
  hipMemcpyAsync(outL, d_in[5], (size_t)133 * LM2 * sizeof(float), hipMemcpyDeviceToDevice, stream);
  hipMemcpyAsync(outG, d_in[6], (size_t)133 * GM2 * sizeof(float), hipMemcpyDeviceToDevice, stream);
  k_init_state<<<1, 1, 0, stream>>>(st, lmbst, (const float*)d_in[7], (const float*)d_in[8],
                                    (const int*)d_in[9], (const float*)d_in[10]);

  const int chunks[3][2] = {{0, 43}, {43, 43}, {86, 43}};
  for (int t = 0; t < 8; ++t) {
    const unsigned char* done = dones + t;
    const unsigned char* upd  = upds + t;
    const float* del   = delta + (size_t)t * 3;
    const float* depth = obs + ((size_t)t * 4 + 3) * (FRAME_H * FRAME_W);
    const float* pixT  = pixF + (size_t)t * FDIM * (FRAME_H * FRAME_W);

    k_reset_maps<<<(unsigned)(((size_t)133 * GM2 + 255) / 256), 256, 0, stream>>>(done, outL, outG);
    k_pose<<<1, 1, 0, stream>>>(done, del, st, lmbst);
    k_pool<<<FDIM, 256, 0, stream>>>(pixT, pooled);

    for (int cc = 0; cc < 3; ++cc) {
      int c0 = chunks[cc][0], nc = chunks[cc][1];
      hipMemsetAsync(vox, 0, (size_t)nc * VRR * VRR * NVH * sizeof(float), stream);
      int wpp = (nc + 31) >> 5;
      int thr = NPTS * wpp * 32;
      k_splat<<<(thr + 255) / 256, 256, 0, stream>>>(depth, pooled, vox, c0, nc);
      int nred = nc * VRR * VRR;
      k_reduce<<<(nred + 255) / 256, 256, 0, stream>>>(vox, agent, all0, c0, nc);
    }

    k_av<<<(unsigned)(((size_t)NMAPC * LM2 + 255) / 256), 256, 0, stream>>>(agent, all0, av);
    dim3 rg(LOCALM / 16, LOCALM / 16, NMAPC);
    k_rot<<<rg, 256, 0, stream>>>(av, st, rotb);
    k_trn<<<(unsigned)(((size_t)NMAPC * LM2 + 255) / 256), 256, 0, stream>>>(rotb, st, outL);

    k_gwrite<<<(unsigned)(((size_t)NMAPC * LM2 + 255) / 256), 256, 0, stream>>>(upd, lmbst, outL, outG);
    k_recenter<<<1, 1, 0, stream>>>(upd, st, lmbst,
                                    outLP + (size_t)t * 3, outGP + (size_t)t * 3,
                                    outLMB + (size_t)t * 4, outORG + (size_t)t * 3);
    k_lfromg<<<(unsigned)(((size_t)NMAPC * LM2 + 255) / 256), 256, 0, stream>>>(upd, lmbst, outG, outL);

    k_mf<<<(unsigned)(((size_t)136 * LM2 + 255) / 256), 256, 0, stream>>>(
        outL, outG, outMF + (size_t)t * 136 * LM2);
  }
}
